// MultiheadSelfAttention_79474074845309
// MI455X (gfx1250) — compile-verified, matrix operand order checked by reading, not run on hardware
//
#include <hip/hip_runtime.h>

#define D_MODEL   1024
#define NUM_HEADS 16
#define HEAD_DIM  64
#define SEQ       2048
#define BATCH     2

typedef __attribute__((ext_vector_type(16))) __bf16 v16bf;
typedef __attribute__((ext_vector_type(8)))  __bf16 v8bf;
typedef __attribute__((ext_vector_type(4)))  __bf16 v4bf;
typedef __attribute__((ext_vector_type(8)))  float  v8f;

#define WMMA_BF16(a, b, c) \
  __builtin_amdgcn_wmma_f32_16x16x32_bf16(false, (a), false, (b), (short)0, (c), false, false)

// ln(10000)/64 : RoPE inv_freq = exp(-(d&~1) * LN_THETA_OVER_DK)
#define LN_THETA_OVER_DK 0.143911568312128f

// ---------------------------------------------------------------------------
// WMMA 16-bit A/B operand loader (wave32). Lane L holds row/col (L%16),
// group g = L/16; element j maps to contraction index
//   K = (j>>3)*16 + g*8 + (j&7)
// i.e. two contiguous 8-element (16 B) runs -> two b128 loads, no conversion.
// ---------------------------------------------------------------------------
__device__ __forceinline__ v16bf load_row16(const __bf16* __restrict__ row, int group) {
  v8bf lo = *reinterpret_cast<const v8bf*>(row + group * 8);
  v8bf hi = *reinterpret_cast<const v8bf*>(row + 16 + group * 8);
  return __builtin_shufflevector(lo, hi, 0, 1, 2, 3, 4, 5, 6, 7,
                                          8, 9, 10, 11, 12, 13, 14, 15);
}

// ---------------------------------------------------------------------------
// One-time f32 -> bf16 conversion (memory-bound pre-pass; negligible at
// 23.3 TB/s next to the compute-bound GEMMs it feeds).
// ---------------------------------------------------------------------------
__global__ void f32_to_bf16_kernel(const float* __restrict__ in,
                                   __bf16* __restrict__ out, int n4) {
  int i = blockIdx.x * blockDim.x + threadIdx.x;
  if (i < n4) {
    float4 v = reinterpret_cast<const float4*>(in)[i];
    v4bf o;
    o[0] = (__bf16)v.x; o[1] = (__bf16)v.y; o[2] = (__bf16)v.z; o[3] = (__bf16)v.w;
    reinterpret_cast<v4bf*>(out)[i] = o;
  }
}

// ---------------------------------------------------------------------------
// Shared GEMM mainloop: C[32x64] per wave = A[rows,K] x W[cols,K]^T, bf16 in,
// f32 accumulate. 12 b128 loads + 8 wmma per K-step; operands are L2-resident.
// ---------------------------------------------------------------------------
__device__ __forceinline__ void gemm_mainloop(const __bf16* __restrict__ A,
                                              const __bf16* __restrict__ W,
                                              int K, int row0, int col0,
                                              int m_in, int group, v8f acc[2][4]) {
  for (int k0 = 0; k0 < K; k0 += 32) {
    v16bf a0 = load_row16(A + (size_t)(row0 + m_in) * K + k0, group);
    v16bf a1 = load_row16(A + (size_t)(row0 + 16 + m_in) * K + k0, group);
    v16bf bm[4];
#pragma unroll
    for (int nt = 0; nt < 4; ++nt)
      bm[nt] = load_row16(W + (size_t)(col0 + nt * 16 + m_in) * K + k0, group);
#pragma unroll
    for (int nt = 0; nt < 4; ++nt) {
      acc[0][nt] = WMMA_BF16(a0, bm[nt], acc[0][nt]);
      acc[1][nt] = WMMA_BF16(a1, bm[nt], acc[1][nt]);
    }
  }
}

// ---------------------------------------------------------------------------
// Kernel 1: qkv = x @ w_qkv^T, fused RoPE on q/k.
// q,k stored bf16 [b][h][s][64]; v stored TRANSPOSED bf16 [b][h][64][s] so the
// attention PV matmul can use contiguous B-operand loads along the key dim.
// Block = 128 thr (4 waves) -> 64(M) x 128(N) tile. Grid: 24 x 64.
// ---------------------------------------------------------------------------
__global__ void qkv_rope_kernel(const __bf16* __restrict__ xb,
                                const __bf16* __restrict__ wqkvb,
                                const int* __restrict__ pos,
                                __bf16* __restrict__ qws,
                                __bf16* __restrict__ kws,
                                __bf16* __restrict__ vtws) {
  const int lane  = threadIdx.x & 31;
  const int wv    = threadIdx.x >> 5;
  const int m_in  = lane & 15;
  const int group = lane >> 4;
  const int row0  = blockIdx.y * 64 + (wv >> 1) * 32;
  const int col0  = blockIdx.x * 128 + (wv & 1) * 64;

  v8f acc[2][4] = {};
  gemm_mainloop(xb, wqkvb, D_MODEL, row0, col0, m_in, group, acc);

#pragma unroll
  for (int mt = 0; mt < 2; ++mt) {
#pragma unroll
    for (int nt = 0; nt < 4; ++nt) {
      const int e    = col0 + nt * 16 + m_in;   // output feature (lane = N col)
      const int sect = e >> 10;                 // 0=q, 1=k, 2=v (wave-uniform)
      const int ei   = e & 1023;
      const int h    = ei >> 6;
      const int d    = ei & 63;
      // lane-uniform RoPE frequency, hoisted out of the row loop:
      // theta^(-(d&~1)/dk) = exp(-(d&~1) * ln(theta)/dk)  -> one v_exp_f32
      const float freq = __expf(-(float)(d & ~1) * LN_THETA_OVER_DK);
#pragma unroll
      for (int r = 0; r < 8; ++r) {
        const int row = row0 + mt * 16 + 8 * group + r;  // C layout: m = 8*g + r
        const int bb  = row >> 11;
        const int ss  = row & 2047;
        const int bh  = bb * NUM_HEADS + h;
        float v = acc[mt][nt][r];
        float partner = __shfl_xor(v, 1);                // RoPE pair column d^1
        if (sect < 2) {
          float ang = (float)pos[ss] * freq;
          float sn, cs;
          __sincosf(ang, &sn, &cs);
          v = v * cs + ((d & 1) ? partner : -partner) * sn;
          __bf16* dst = (sect == 0) ? qws : kws;
          dst[((size_t)bh * SEQ + ss) * HEAD_DIM + d] = (__bf16)v;
        } else {
          vtws[((size_t)bh * HEAD_DIM + d) * SEQ + ss] = (__bf16)v;  // V^T
        }
      }
    }
  }
}

// half-wave (16-lane) reductions: C-layout rows live in one 16-lane half
__device__ __forceinline__ float halfmax(float v) {
  v = fmaxf(v, __shfl_xor(v, 1));
  v = fmaxf(v, __shfl_xor(v, 2));
  v = fmaxf(v, __shfl_xor(v, 4));
  v = fmaxf(v, __shfl_xor(v, 8));
  return v;
}
__device__ __forceinline__ float halfsum(float v) {
  v += __shfl_xor(v, 1);
  v += __shfl_xor(v, 2);
  v += __shfl_xor(v, 4);
  v += __shfl_xor(v, 8);
  return v;
}

// ---------------------------------------------------------------------------
// Kernel 2: causal flash attention. 4 waves/block, one 16-row q block each.
// Grid: x = 32 (q-block groups of 4), y = 32 (b*h).
// ---------------------------------------------------------------------------
__global__ void attn_kernel(const __bf16* __restrict__ qws,
                            const __bf16* __restrict__ kws,
                            const __bf16* __restrict__ vtws,
                            __bf16* __restrict__ ows) {
  __shared__ __align__(16) __bf16 lds_p[4][16 * 32];

  const int lane  = threadIdx.x & 31;
  const int wv    = threadIdx.x >> 5;
  const int n     = lane & 15;
  const int group = lane >> 4;
  const int bh    = blockIdx.y;
  const int b     = bh >> 4;
  const int h     = bh & 15;
  const int q0    = (blockIdx.x * 4 + wv) * 16;

  const __bf16* Q  = qws  + (size_t)bh * SEQ * HEAD_DIM;
  const __bf16* K  = kws  + (size_t)bh * SEQ * HEAD_DIM;
  const __bf16* Vt = vtws + (size_t)bh * HEAD_DIM * SEQ;   // [64][2048]

  // Q tile (16 x 64) as two A operands (K split 0..31 / 32..63)
  v16bf aQ0 = load_row16(Q + (size_t)(q0 + n) * HEAD_DIM, group);
  v16bf aQ1 = load_row16(Q + (size_t)(q0 + n) * HEAD_DIM + 32, group);

  float mrow[8], lrow[8], alpha[8];
  v8f o[4] = {};
#pragma unroll
  for (int r = 0; r < 8; ++r) { mrow[r] = -1e30f; lrow[r] = 0.0f; }

  const int jend = (q0 + 16 + 31) >> 5;  // causal: keys 0 .. q0+15
  for (int j = 0; j < jend; ++j) {
    const int kb = j * 32;

    // scores: S = Q K^T, two 16-wide key tiles, contraction over dk=64
    v8f s0 = {}, s1 = {};
    {
      v16bf b0 = load_row16(K + (size_t)(kb + n) * HEAD_DIM, group);
      v16bf b1 = load_row16(K + (size_t)(kb + n) * HEAD_DIM + 32, group);
      s0 = WMMA_BF16(aQ0, b0, s0);
      s0 = WMMA_BF16(aQ1, b1, s0);
    }
    {
      v16bf b0 = load_row16(K + (size_t)(kb + 16 + n) * HEAD_DIM, group);
      v16bf b1 = load_row16(K + (size_t)(kb + 16 + n) * HEAD_DIM + 32, group);
      s1 = WMMA_BF16(aQ0, b0, s1);
      s1 = WMMA_BF16(aQ1, b1, s1);
    }

    // online softmax (scale 1/sqrt(64) = 0.125), causal mask
#pragma unroll
    for (int r = 0; r < 8; ++r) {
      const int qrow = q0 + 8 * group + r;
      float v0 = (kb + n      <= qrow) ? s0[r] * 0.125f : -1e30f;
      float v1 = (kb + 16 + n <= qrow) ? s1[r] * 0.125f : -1e30f;
      float mx   = halfmax(fmaxf(v0, v1));
      float mnew = fmaxf(mrow[r], mx);
      float a    = __expf(mrow[r] - mnew);
      float p0   = __expf(v0 - mnew);
      float p1   = __expf(v1 - mnew);
      lrow[r]  = lrow[r] * a + halfsum(p0 + p1);
      mrow[r]  = mnew;
      alpha[r] = a;
      s0[r] = p0;
      s1[r] = p1;
    }
#pragma unroll
    for (int dt = 0; dt < 4; ++dt)
#pragma unroll
      for (int r = 0; r < 8; ++r) o[dt][r] *= alpha[r];

    // P: C-layout -> A-layout via per-wave LDS slab (intra-wave DS ordering;
    // cross-lane visibility needs only s_wait_dscnt, no barrier)
    __bf16* P = &lds_p[wv][0];
#pragma unroll
    for (int r = 0; r < 8; ++r) {
      const int rowi = 8 * group + r;
      P[rowi * 32 + n]      = (__bf16)s0[r];
      P[rowi * 32 + 16 + n] = (__bf16)s1[r];
    }
    asm volatile("s_wait_dscnt 0x0" ::: "memory");
    v16bf aP = load_row16((const __bf16*)P + n * 32, group);

    // O += P V : with V^T resident, the B operand (contraction = key dim) is
    // two contiguous b128 loads per d-tile
#pragma unroll
    for (int dt = 0; dt < 4; ++dt) {
      v16bf bV = load_row16(Vt + (size_t)(dt * 16 + n) * SEQ + kb, group);
      o[dt] = WMMA_BF16(aP, bV, o[dt]);
    }
  }

  // normalize (one reciprocal per row), store merged-head bf16 [b][s][1024]
#pragma unroll
  for (int r = 0; r < 8; ++r) {
    const int ss = q0 + 8 * group + r;
    const float rinv = 1.0f / lrow[r];
#pragma unroll
    for (int dt = 0; dt < 4; ++dt) {
      ows[((size_t)b * SEQ + ss) * D_MODEL + h * HEAD_DIM + dt * 16 + n] =
          (__bf16)(o[dt][r] * rinv);
    }
  }
}

// ---------------------------------------------------------------------------
// Kernel 3: out = attn_out @ w_o^T  (M=4096, N=1024, K=1024), f32 output.
// ---------------------------------------------------------------------------
__global__ void out_proj_kernel(const __bf16* __restrict__ aws,
                                const __bf16* __restrict__ wob,
                                float* __restrict__ out) {
  const int lane  = threadIdx.x & 31;
  const int wv    = threadIdx.x >> 5;
  const int m_in  = lane & 15;
  const int group = lane >> 4;
  const int row0  = blockIdx.y * 64 + (wv >> 1) * 32;
  const int col0  = blockIdx.x * 128 + (wv & 1) * 64;

  v8f acc[2][4] = {};
  gemm_mainloop(aws, wob, D_MODEL, row0, col0, m_in, group, acc);

#pragma unroll
  for (int mt = 0; mt < 2; ++mt)
#pragma unroll
    for (int nt = 0; nt < 4; ++nt) {
      const int e = col0 + nt * 16 + m_in;
#pragma unroll
      for (int r = 0; r < 8; ++r) {
        const int row = row0 + mt * 16 + 8 * group + r;
        out[(size_t)row * D_MODEL + e] = acc[mt][nt][r];
      }
    }
}

extern "C" void kernel_launch(void* const* d_in, const int* in_sizes, int n_in,
                              void* d_out, int out_size, void* d_ws, size_t ws_size,
                              hipStream_t stream) {
  const float* x     = (const float*)d_in[0];
  const int*   pos   = (const int*)d_in[1];
  const float* w_qkv = (const float*)d_in[2];
  const float* w_o   = (const float*)d_in[3];
  float* out = (float*)d_out;

  const size_t n_x    = (size_t)BATCH * SEQ * D_MODEL;       // 4 M
  const size_t n_wqkv = (size_t)3 * D_MODEL * D_MODEL;       // 3 M
  const size_t n_wo   = (size_t)D_MODEL * D_MODEL;           // 1 M
  const size_t n_hd   = (size_t)BATCH * NUM_HEADS * SEQ * HEAD_DIM;  // 4 M

  __bf16* xb    = (__bf16*)d_ws;          // [4096][1024]
  __bf16* wqkvb = xb    + n_x;            // [3072][1024]
  __bf16* wob   = wqkvb + n_wqkv;         // [1024][1024]
  __bf16* qws   = wob   + n_wo;           // [b][h][s][64]
  __bf16* kws   = qws   + n_hd;           // [b][h][s][64]
  __bf16* vtws  = kws   + n_hd;           // [b][h][64][s]  (transposed)
  __bf16* ows   = vtws  + n_hd;           // [b][s][1024]   -- 48 MB total

  // one-time bf16 conversion of GEMM operands
  f32_to_bf16_kernel<<<(int)(n_x    / 1024), 256, 0, stream>>>(x,     xb,    (int)(n_x    / 4));
  f32_to_bf16_kernel<<<(int)(n_wqkv / 1024), 256, 0, stream>>>(w_qkv, wqkvb, (int)(n_wqkv / 4));
  f32_to_bf16_kernel<<<(int)(n_wo   / 1024), 256, 0, stream>>>(w_o,   wob,   (int)(n_wo   / 4));

  // QKV projection + RoPE: N=3072 -> grid.x = 24, M=4096 -> grid.y = 64
  qkv_rope_kernel<<<dim3(24, 64), 128, 0, stream>>>(xb, wqkvb, pos, qws, kws, vtws);
  // attention: 32 q-block groups x 32 (b,h) pairs
  attn_kernel<<<dim3(32, 32), 128, 0, stream>>>(qws, kws, vtws, ows);
  // output projection: N=1024 -> grid.x = 8
  out_proj_kernel<<<dim3(8, 64), 128, 0, stream>>>(ows, wob, out);
}
